// ResidualBlock_80307298500857
// MI455X (gfx1250) — matrix operand order repaired and checked by reading, not compiled, not run on hardware
//
#include <hip/hip_runtime.h>
#include <hip/hip_bf16.h>

typedef __bf16 bf16;
typedef __attribute__((ext_vector_type(16))) __bf16        bf16x16;
typedef __attribute__((ext_vector_type(8)))  float         f32x8;
typedef __attribute__((ext_vector_type(4)))  unsigned int  u32x4;
typedef __attribute__((ext_vector_type(8)))  unsigned int  u32x8;

#define D_MODEL   768
#define N_Q       12
#define Q_DIM     64
#define WINDOW    128
#define MLP_INNER 3072
#define SEQ       2048
#define QKV_COLS  (Q_DIM * (N_Q + 2))   // 896

// ---------------- helpers ----------------
__device__ __forceinline__ bf16x16 make_frag(u32x4 lo, u32x4 hi) {
  u32x8 w = { lo[0], lo[1], lo[2], lo[3], hi[0], hi[1], hi[2], hi[3] };
  return __builtin_bit_cast(bf16x16, w);
}

__device__ __forceinline__ float gelu_tanh(float x) {
  const float c = 0.7978845608028654f; // sqrt(2/pi)
  float x3 = x * x * x;
  return 0.5f * x * (1.0f + tanhf(c * (x + 0.044715f * x3)));
}

// Low 32 bits of a generic pointer to __shared__ == LDS byte address.
__device__ __forceinline__ unsigned lds_off(const void* p) {
  return (unsigned)(unsigned long long)p;
}

// Async copy 16B global -> LDS (tracked by ASYNCcnt, no VGPR staging).
__device__ __forceinline__ void async_ld_b128(unsigned ldsaddr, const void* gptr) {
  asm volatile("global_load_async_to_lds_b128 %0, %1, off"
               :: "v"(ldsaddr), "v"(gptr) : "memory");
}

__device__ __forceinline__ void wait_async0() {
  asm volatile("s_wait_asynccnt 0x0" ::: "memory");
}

// ---------------- fp32 -> bf16 cast ----------------
__global__ void cast_f32_bf16(const float* __restrict__ in, bf16* __restrict__ out, int n) {
  int i = blockIdx.x * blockDim.x + threadIdx.x;
  int stride = gridDim.x * blockDim.x;
  for (; i < n; i += stride) out[i] = (bf16)in[i];
}

// ---------------- WMMA GEMM: C[M,N] = A[M,K](bf16) @ B[N,K](bf16)^T + bias (+ resid) ----------------
#define BM 128
#define BN 64
#define BK 32
#define LDSTR 40   // padded LDS row stride (bf16): 80B rows keep 16B alignment, break bank repeats

template <bool HAS_RESID>
__global__ __launch_bounds__(256)
void gemm_bias_resid(const bf16* __restrict__ A, const bf16* __restrict__ B,
                     const float* __restrict__ bias, const float* __restrict__ resid,
                     float* __restrict__ C, int M, int N, int K)
{
  __shared__ __align__(16) bf16 As[2][BM * LDSTR];
  __shared__ __align__(16) bf16 Bs[2][BN * LDSTR];
  const int tid  = threadIdx.x;
  const int lane = tid & 31;
  const int wave = tid >> 5;
  const int wm = wave & 3;    // 4 waves along M (32 rows each)
  const int wn = wave >> 2;   // 2 waves along N (32 cols each)
  const int bm = blockIdx.y * BM;
  const int bn = blockIdx.x * BN;

  f32x8 acc[2][2] = {};

  const int ar = tid >> 1;          // A tile: row 0..127
  const int ac = (tid & 1) * 16;    //         col 0 or 16 (16 bf16 per thread)
  const int br = tid >> 2;          // B tile: row 0..63
  const int bc = (tid & 3) * 8;     //         col 0/8/16/24 (8 bf16 per thread)
  const int arow = lane & 15;
  const int akb  = (lane < 16) ? 0 : 8;    // A frag K-base (16-bit A 16x32 layout)
  const int bkb  = (lane < 16) ? 0 : 8;    // B frag K-base (16-bit B 32x16 layout)

  const bf16* aptr = A + (size_t)(bm + ar) * K + ac;
  const bf16* bptr = B + (size_t)(bn + br) * K + bc;
  unsigned lA[2], lB[2];
  lA[0] = lds_off(&As[0][ar * LDSTR + ac]);
  lA[1] = lds_off(&As[1][ar * LDSTR + ac]);
  lB[0] = lds_off(&Bs[0][br * LDSTR + bc]);
  lB[1] = lds_off(&Bs[1][br * LDSTR + bc]);

  const int nk = K / BK;
  // prologue: async-load tile 0 into buffer 0
  async_ld_b128(lA[0],      aptr);
  async_ld_b128(lA[0] + 16, aptr + 8);
  async_ld_b128(lB[0],      bptr);
  wait_async0();
  __syncthreads();

  for (int k = 0; k < nk; ++k) {
    const int p = k & 1;
    if (k + 1 < nk) {  // issue next tile into the other buffer; overlaps with compute
      const bf16* an = aptr + (size_t)(k + 1) * BK;
      const bf16* bn2 = bptr + (size_t)(k + 1) * BK;
      async_ld_b128(lA[p ^ 1],      an);
      async_ld_b128(lA[p ^ 1] + 16, an + 8);
      async_ld_b128(lB[p ^ 1],      bn2);
    }

    bf16x16 afrag[2], bfrag[2];
#pragma unroll
    for (int i = 0; i < 2; ++i) {
      int r = wm * 32 + i * 16 + arow;
      afrag[i] = make_frag(*(const u32x4*)&As[p][r * LDSTR + akb],
                           *(const u32x4*)&As[p][r * LDSTR + akb + 16]);
    }
#pragma unroll
    for (int j = 0; j < 2; ++j) {
      int cix = wn * 32 + j * 16 + arow;
      bfrag[j] = make_frag(*(const u32x4*)&Bs[p][cix * LDSTR + bkb],
                           *(const u32x4*)&Bs[p][cix * LDSTR + bkb + 16]);
    }
#pragma unroll
    for (int i = 0; i < 2; ++i)
#pragma unroll
      for (int j = 0; j < 2; ++j)
        acc[i][j] = __builtin_amdgcn_wmma_f32_16x16x32_bf16(
            false, afrag[i], false, bfrag[j], (short)0, acc[i][j], false, false);

    if (k + 1 < nk) { wait_async0(); __syncthreads(); }
  }

#pragma unroll
  for (int i = 0; i < 2; ++i) {
#pragma unroll
    for (int j = 0; j < 2; ++j) {
      int col = bn + wn * 32 + j * 16 + (lane & 15);
      float bv = bias[col];
      int rb = bm + wm * 32 + i * 16 + ((lane < 16) ? 0 : 8);
      size_t idx = (size_t)rb * N + col;
#pragma unroll
      for (int r = 0; r < 8; ++r) {
        float v = acc[i][j][r] + bv;
        if (HAS_RESID) v += resid[idx];
        C[idx] = v;
        idx += N;
      }
    }
  }
}

// ---------------- fused GLU GEMM: H = gelu(xn@W1a^T+b1a) * (xn@W1b^T+b1b), bf16 out ----------------
__global__ __launch_bounds__(256)
void glu_gemm(const bf16* __restrict__ A /*[M,K]*/, const bf16* __restrict__ B /*[2*MLP_INNER,K]*/,
              const float* __restrict__ bias /*[2*MLP_INNER]*/, bf16* __restrict__ H /*[M,MLP_INNER]*/,
              int M, int K)
{
  __shared__ __align__(16) bf16 As[2][BM * LDSTR];
  __shared__ __align__(16) bf16 Bsa[2][BN * LDSTR];
  __shared__ __align__(16) bf16 Bsb[2][BN * LDSTR];
  const int tid  = threadIdx.x;
  const int lane = tid & 31;
  const int wave = tid >> 5;
  const int wm = wave & 3;
  const int wn = wave >> 2;
  const int bm = blockIdx.y * BM;
  const int bn = blockIdx.x * BN;   // column base in [0, MLP_INNER)

  f32x8 acca[2][2] = {};
  f32x8 accb[2][2] = {};

  const int ar = tid >> 1;
  const int ac = (tid & 1) * 16;
  const int br = tid >> 2;
  const int bc = (tid & 3) * 8;
  const int arow = lane & 15;
  const int akb  = (lane < 16) ? 0 : 8;
  const int bkb  = (lane < 16) ? 0 : 8;

  const bf16* aptr  = A + (size_t)(bm + ar) * K + ac;
  const bf16* bptrA = B + (size_t)(bn + br) * K + bc;
  const bf16* bptrB = B + (size_t)(bn + br + MLP_INNER) * K + bc;
  unsigned lA[2], lBa[2], lBb[2];
  lA[0]  = lds_off(&As[0][ar * LDSTR + ac]);
  lA[1]  = lds_off(&As[1][ar * LDSTR + ac]);
  lBa[0] = lds_off(&Bsa[0][br * LDSTR + bc]);
  lBa[1] = lds_off(&Bsa[1][br * LDSTR + bc]);
  lBb[0] = lds_off(&Bsb[0][br * LDSTR + bc]);
  lBb[1] = lds_off(&Bsb[1][br * LDSTR + bc]);

  const int nk = K / BK;
  async_ld_b128(lA[0],      aptr);
  async_ld_b128(lA[0] + 16, aptr + 8);
  async_ld_b128(lBa[0],     bptrA);
  async_ld_b128(lBb[0],     bptrB);
  wait_async0();
  __syncthreads();

  for (int k = 0; k < nk; ++k) {
    const int p = k & 1;
    if (k + 1 < nk) {
      size_t koff = (size_t)(k + 1) * BK;
      async_ld_b128(lA[p ^ 1],      aptr + koff);
      async_ld_b128(lA[p ^ 1] + 16, aptr + koff + 8);
      async_ld_b128(lBa[p ^ 1],     bptrA + koff);
      async_ld_b128(lBb[p ^ 1],     bptrB + koff);
    }

    bf16x16 afrag[2], bfa[2], bfb[2];
#pragma unroll
    for (int i = 0; i < 2; ++i) {
      int r = wm * 32 + i * 16 + arow;
      afrag[i] = make_frag(*(const u32x4*)&As[p][r * LDSTR + akb],
                           *(const u32x4*)&As[p][r * LDSTR + akb + 16]);
    }
#pragma unroll
    for (int j = 0; j < 2; ++j) {
      int cix = wn * 32 + j * 16 + arow;
      bfa[j] = make_frag(*(const u32x4*)&Bsa[p][cix * LDSTR + bkb],
                         *(const u32x4*)&Bsa[p][cix * LDSTR + bkb + 16]);
      bfb[j] = make_frag(*(const u32x4*)&Bsb[p][cix * LDSTR + bkb],
                         *(const u32x4*)&Bsb[p][cix * LDSTR + bkb + 16]);
    }
#pragma unroll
    for (int i = 0; i < 2; ++i)
#pragma unroll
      for (int j = 0; j < 2; ++j) {
        acca[i][j] = __builtin_amdgcn_wmma_f32_16x16x32_bf16(
            false, afrag[i], false, bfa[j], (short)0, acca[i][j], false, false);
        accb[i][j] = __builtin_amdgcn_wmma_f32_16x16x32_bf16(
            false, afrag[i], false, bfb[j], (short)0, accb[i][j], false, false);
      }

    if (k + 1 < nk) { wait_async0(); __syncthreads(); }
  }

#pragma unroll
  for (int i = 0; i < 2; ++i) {
#pragma unroll
    for (int j = 0; j < 2; ++j) {
      int col = bn + wn * 32 + j * 16 + (lane & 15);
      float ba = bias[col];
      float bb = bias[col + MLP_INNER];
      int rb = bm + wm * 32 + i * 16 + ((lane < 16) ? 0 : 8);
      size_t idx = (size_t)rb * MLP_INNER + col;
#pragma unroll
      for (int r = 0; r < 8; ++r) {
        float a = acca[i][j][r] + ba;
        float b = accb[i][j][r] + bb;
        H[idx] = (bf16)(gelu_tanh(a) * b);
        idx += MLP_INNER;
      }
    }
  }
}

// ---------------- windowed MQA attention (~2% of FLOPs; scalar, LDS-staged) ----------------
__global__ __launch_bounds__(256)
void attn_kernel(const float* __restrict__ qkv /*[SEQ,896]: k|v|q*/, bf16* __restrict__ out /*[SEQ,768]*/)
{
  __shared__ float qh[N_Q * Q_DIM];       // 3 KB
  __shared__ float kv[WINDOW * Q_DIM];    // 32 KB, staged K then reused for V
  __shared__ float sc[N_Q * WINDOW];      // 6 KB
  const int s   = blockIdx.x;
  const int tid = threadIdx.x;
  const int T   = min(s + 1, WINDOW);
  const int t0  = s + 1 - T;

  const float* qrow = qkv + (size_t)s * QKV_COLS + 2 * Q_DIM;
  for (int i = tid; i < N_Q * Q_DIM; i += 256) qh[i] = qrow[i];
  for (int i = tid; i < T * Q_DIM; i += 256) {
    int tt = i >> 6, d = i & 63;
    kv[i] = qkv[(size_t)(t0 + tt) * QKV_COLS + d];   // K
  }
  __syncthreads();

  for (int p = tid; p < N_Q * T; p += 256) {
    int h = p / T, tt = p - h * T;
    const float* qp = qh + h * Q_DIM;
    const float* kp = kv + tt * Q_DIM;
    float acc = 0.0f;
#pragma unroll 8
    for (int d = 0; d < Q_DIM; ++d) acc += qp[d] * kp[d];
    sc[h * WINDOW + tt] = 8.0f * acc;   // sqrt(Q_DIM) * q.k (reference multiplies)
  }
  __syncthreads();

  const int lane = tid & 31, wave = tid >> 5;
  for (int h = wave; h < N_Q; h += 8) {
    float mx = -3.0e38f;
    for (int tt = lane; tt < T; tt += 32) mx = fmaxf(mx, sc[h * WINDOW + tt]);
    for (int o = 16; o > 0; o >>= 1) mx = fmaxf(mx, __shfl_xor(mx, o, 32));
    float sum = 0.0f;
    for (int tt = lane; tt < T; tt += 32) {
      float e = __expf(sc[h * WINDOW + tt] - mx);
      sc[h * WINDOW + tt] = e;
      sum += e;
    }
    for (int o = 16; o > 0; o >>= 1) sum += __shfl_xor(sum, o, 32);
    float inv = 1.0f / sum;
    for (int tt = lane; tt < T; tt += 32) sc[h * WINDOW + tt] *= inv;
  }
  __syncthreads();

  for (int i = tid; i < T * Q_DIM; i += 256) {
    int tt = i >> 6, d = i & 63;
    kv[i] = qkv[(size_t)(t0 + tt) * QKV_COLS + Q_DIM + d];   // V
  }
  __syncthreads();

  for (int o = tid; o < N_Q * Q_DIM; o += 256) {
    int h = o >> 6, d = o & 63;
    const float* sp = sc + h * WINDOW;
    float acc = 0.0f;
    for (int tt = 0; tt < T; ++tt) acc += sp[tt] * kv[tt * Q_DIM + d];
    out[(size_t)s * (N_Q * Q_DIM) + o] = (bf16)acc;
  }
}

// ---------------- RMSNorm -> bf16 ----------------
__global__ __launch_bounds__(256)
void rmsnorm_kernel(const float* __restrict__ x, const float* __restrict__ g, bf16* __restrict__ out)
{
  __shared__ float red[8];
  const int row = blockIdx.x;
  const float* xr = x + (size_t)row * D_MODEL;
  float s = 0.0f;
  for (int i = threadIdx.x; i < D_MODEL; i += 256) { float v = xr[i]; s += v * v; }
  for (int o = 16; o > 0; o >>= 1) s += __shfl_xor(s, o, 32);
  const int lane = threadIdx.x & 31, wave = threadIdx.x >> 5;
  if (lane == 0) red[wave] = s;
  __syncthreads();
  if (threadIdx.x == 0) {
    float t = 0.0f;
    for (int w = 0; w < 8; ++w) t += red[w];
    red[0] = rsqrtf(t / (float)D_MODEL + 1e-5f);
  }
  __syncthreads();
  float r = red[0];
  for (int i = threadIdx.x; i < D_MODEL; i += 256)
    out[(size_t)row * D_MODEL + i] = (bf16)(xr[i] * r * g[i]);
}

// ---------------- launch ----------------
extern "C" void kernel_launch(void* const* d_in, const int* in_sizes, int n_in,
                              void* d_out, int out_size, void* d_ws, size_t ws_size,
                              hipStream_t stream) {
  const float* x    = (const float*)d_in[0];
  // d_in[1] = state (unused by reference)
  const float* Wqkv = (const float*)d_in[2];
  const float* bqkv = (const float*)d_in[3];
  const float* Wout = (const float*)d_in[4];
  const float* bout = (const float*)d_in[5];
  const float* W1   = (const float*)d_in[6];
  const float* b1   = (const float*)d_in[7];
  const float* W2   = (const float*)d_in[8];
  const float* b2   = (const float*)d_in[9];
  const float* g2   = (const float*)d_in[10];
  float* out = (float*)d_out;

  char* ws = (char*)d_ws;
  size_t off = 0;
  auto alloc = [&](size_t bytes) -> void* {
    void* p = ws + off;
    off = (off + bytes + 255) & ~(size_t)255;
    return p;
  };

  bf16*  xb    = (bf16*) alloc((size_t)SEQ * D_MODEL * 2);
  bf16*  wqkvb = (bf16*) alloc((size_t)QKV_COLS * D_MODEL * 2);
  bf16*  woutb = (bf16*) alloc((size_t)D_MODEL * D_MODEL * 2);
  bf16*  w1b   = (bf16*) alloc((size_t)2 * MLP_INNER * D_MODEL * 2);
  bf16*  w2b   = (bf16*) alloc((size_t)D_MODEL * MLP_INNER * 2);
  float* qkv   = (float*)alloc((size_t)SEQ * QKV_COLS * 4);
  bf16*  attnb = (bf16*) alloc((size_t)SEQ * D_MODEL * 2);
  float* x1    = (float*)alloc((size_t)SEQ * D_MODEL * 4);
  bf16*  xnb   = (bf16*) alloc((size_t)SEQ * D_MODEL * 2);
  bf16*  hb    = (bf16*) alloc((size_t)SEQ * MLP_INNER * 2);

  cast_f32_bf16<<<512, 256, 0, stream>>>(x,    xb,    SEQ * D_MODEL);
  cast_f32_bf16<<<512, 256, 0, stream>>>(Wqkv, wqkvb, QKV_COLS * D_MODEL);
  cast_f32_bf16<<<512, 256, 0, stream>>>(Wout, woutb, D_MODEL * D_MODEL);
  cast_f32_bf16<<<1024, 256, 0, stream>>>(W1,  w1b,   2 * MLP_INNER * D_MODEL);
  cast_f32_bf16<<<1024, 256, 0, stream>>>(W2,  w2b,   D_MODEL * MLP_INNER);

  // qkv = x @ Wqkv^T + b_qkv
  gemm_bias_resid<false><<<dim3(QKV_COLS / BN, SEQ / BM), 256, 0, stream>>>(
      xb, wqkvb, bqkv, nullptr, qkv, SEQ, QKV_COLS, D_MODEL);

  // windowed MQA
  attn_kernel<<<SEQ, 256, 0, stream>>>(qkv, attnb);

  // x1 = attn @ Wout^T + b_out + x   (residual fused)
  gemm_bias_resid<true><<<dim3(D_MODEL / BN, SEQ / BM), 256, 0, stream>>>(
      attnb, woutb, bout, x, x1, SEQ, D_MODEL, D_MODEL);

  // xn = rmsnorm(x1) * g2  -> bf16
  rmsnorm_kernel<<<SEQ, 256, 0, stream>>>(x1, g2, xnb);

  // h = gelu(xn@W1a^T+b1a) * (xn@W1b^T+b1b)  -> bf16, never materializes u
  glu_gemm<<<dim3(MLP_INNER / BN, SEQ / BM), 256, 0, stream>>>(
      xnb, w1b, b1, hb, SEQ, D_MODEL);

  // out = h @ W2^T + b2 + x1   (residual fused)
  gemm_bias_resid<true><<<dim3(D_MODEL / BN, SEQ / BM), 256, 0, stream>>>(
      hb, w2b, b2, x1, out, SEQ, D_MODEL, MLP_INNER);
}
